// PerceiverSelfAttention_30064771072508
// MI455X (gfx1250) — matrix operand order repaired and checked by reading, not compiled, not run on hardware
//
#include <hip/hip_runtime.h>
#include <hip/hip_bf16.h>
#include <math.h>
#include <stdint.h>

// Problem dims (fixed by the reference)
#define BB     8
#define QL     512
#define KLEN   4096
#define DMODEL 1024
#define NH     8
#define DH     128

typedef __bf16 v16bf __attribute__((ext_vector_type(16)));
typedef __bf16 v8bf  __attribute__((ext_vector_type(8)));
typedef float  v8f   __attribute__((ext_vector_type(8)));
typedef int    v4i   __attribute__((ext_vector_type(4)));

union FragA { v16bf v; v8bf h[2]; };

__device__ __forceinline__ unsigned short f2bf(float f) {
  union { float f; unsigned u; } c; c.f = f;
  unsigned r = c.u + 0x7FFFu + ((c.u >> 16) & 1u);  // round-to-nearest-even
  return (unsigned short)(r >> 16);
}

__device__ __forceinline__ v8f wmma_bf16(const FragA& a, const FragA& b, v8f c) {
  return __builtin_amdgcn_wmma_f32_16x16x32_bf16(
      /*neg_a=*/false, a.v, /*neg_b=*/false, b.v,
      /*c_mod=*/(short)0, c, /*reuse_a=*/false, /*reuse_b=*/false);
}

// ---------------------------------------------------------------------------
// CDNA5 async global->LDS (ASYNCcnt-tracked). Signature discovered via
// compile probe: (int4 AS1* src, int4 AS3* dst, imm offset, imm cpol).
// Guarded: falls back to a synchronous register bounce if not declared.
// ---------------------------------------------------------------------------
#if __has_builtin(__builtin_amdgcn_global_load_async_to_lds_b128)
#define ASYNC_LDS 1
#endif

__device__ __forceinline__ void async_cp16(unsigned short* lds_dst,
                                           const unsigned short* gsrc) {
#ifdef ASYNC_LDS
  __builtin_amdgcn_global_load_async_to_lds_b128(
      (__attribute__((address_space(1))) v4i*)(uintptr_t)gsrc,
      (__attribute__((address_space(3))) v4i*)(unsigned int)(uintptr_t)lds_dst,
      0, 0);
#else
  *(v8bf*)lds_dst = *(const v8bf*)gsrc;
#endif
}

template <int N>
__device__ __forceinline__ void wait_asynccnt() {
#ifdef ASYNC_LDS
#if __has_builtin(__builtin_amdgcn_s_wait_asynccnt)
  __builtin_amdgcn_s_wait_asynccnt((short)N);
#else
  asm volatile("s_wait_asynccnt %0" ::"i"(N) : "memory");
#endif
#endif
}

// ---------------------------------------------------------------------------
// Kernel 1: LayerNorm(f32) -> bf16, one 1024-wide row per 256-thread block
// ---------------------------------------------------------------------------
__global__ __launch_bounds__(256) void ln_bf16_kernel(
    const float* __restrict__ x, const float* __restrict__ w,
    const float* __restrict__ b, unsigned short* __restrict__ y) {
  const int row = blockIdx.x;
  const int t = threadIdx.x;
  const float4 vv = ((const float4*)(x + (size_t)row * DMODEL))[t];

  float s = vv.x + vv.y + vv.z + vv.w;
#pragma unroll
  for (int off = 16; off; off >>= 1) s += __shfl_xor(s, off, 32);
  __shared__ float red[8];
  if ((t & 31) == 0) red[t >> 5] = s;
  __syncthreads();
  float tot = 0.f;
#pragma unroll
  for (int i = 0; i < 8; ++i) tot += red[i];
  const float mean = tot * (1.0f / DMODEL);

  const float d0 = vv.x - mean, d1 = vv.y - mean, d2 = vv.z - mean, d3 = vv.w - mean;
  float s2 = d0 * d0 + d1 * d1 + d2 * d2 + d3 * d3;
#pragma unroll
  for (int off = 16; off; off >>= 1) s2 += __shfl_xor(s2, off, 32);
  __syncthreads();
  if ((t & 31) == 0) red[t >> 5] = s2;
  __syncthreads();
  float tot2 = 0.f;
#pragma unroll
  for (int i = 0; i < 8; ++i) tot2 += red[i];
  const float rs = rsqrtf(tot2 * (1.0f / DMODEL) + 1e-5f);

  const int c0 = t * 4;
  unsigned short* yr = y + (size_t)row * DMODEL + c0;
  yr[0] = f2bf(d0 * rs * w[c0 + 0] + b[c0 + 0]);
  yr[1] = f2bf(d1 * rs * w[c0 + 1] + b[c0 + 1]);
  yr[2] = f2bf(d2 * rs * w[c0 + 2] + b[c0 + 2]);
  yr[3] = f2bf(d3 * rs * w[c0 + 3] + b[c0 + 3]);
}

// ---------------------------------------------------------------------------
// Kernel 2: f32 -> bf16 weight conversion
// ---------------------------------------------------------------------------
__global__ __launch_bounds__(256) void f32_to_bf16_kernel(
    const float* __restrict__ x, unsigned short* __restrict__ y, int n) {
  const int i = blockIdx.x * 256 + threadIdx.x;
  if (i < n) y[i] = f2bf(x[i]);
}

// ---------------------------------------------------------------------------
// Kernel 3: projection GEMM  C[m,n] = sum_d A[m,d]*W[n,d] + bias[n]
// 256 threads = 8 waves; block tile 128(M) x 128(N); wave tile 32 x 64
// (2 A-frags x 4 B-frags -> 8 WMMA per 12 B128 loads per k-step).
// MODE 0/1: out[((b*H+h)*seqLen + s)*128 + d]   (q / k layout)
// MODE 2:   out[((b*H+h)*128 + d)*seqLen + s]   (v transposed layout)
// ---------------------------------------------------------------------------
template <int SEQSHIFT, int MODE>
__global__ __launch_bounds__(256) void proj_gemm_kernel(
    const unsigned short* __restrict__ A, const unsigned short* __restrict__ W,
    const float* __restrict__ bias, unsigned short* __restrict__ out) {
  const int lane = threadIdx.x & 31;
  const int wave = threadIdx.x >> 5;
  const int half = lane >> 4;
  const int c    = lane & 15;
  const int m0 = blockIdx.x * 128 + (wave >> 1) * 32;
  const int n0 = blockIdx.y * 128 + (wave & 1) * 64;

  v8f acc[2][4];
#pragma unroll
  for (int ms = 0; ms < 2; ++ms)
#pragma unroll
    for (int t = 0; t < 4; ++t)
#pragma unroll
      for (int j = 0; j < 8; ++j) acc[ms][t][j] = 0.f;

  // A-fragment: lane holds row m0+c; K chunks at {half*8, 16+half*8}
  const unsigned short* aRow0 = A + (size_t)(m0 + c) * DMODEL + half * 8;
  const unsigned short* aRow1 = aRow0 + (size_t)16 * DMODEL;

  for (int kk = 0; kk < DMODEL; kk += 32) {
    FragA a0, a1;
    a0.h[0] = *(const v8bf*)(aRow0 + kk);
    a0.h[1] = *(const v8bf*)(aRow0 + kk + 16);
    a1.h[0] = *(const v8bf*)(aRow1 + kk);
    a1.h[1] = *(const v8bf*)(aRow1 + kk + 16);
#pragma unroll
    for (int t = 0; t < 4; ++t) {
      // B-fragment: lane holds column n (= W row n), K chunks at {half*8, 16+half*8}
      const unsigned short* wp =
          W + (size_t)(n0 + 16 * t + c) * DMODEL + kk + half * 8;
      FragA bm;
      bm.h[0] = *(const v8bf*)wp;
      bm.h[1] = *(const v8bf*)(wp + 16);
      acc[0][t] = wmma_bf16(a0, bm, acc[0][t]);
      acc[1][t] = wmma_bf16(a1, bm, acc[1][t]);
    }
  }

#pragma unroll
  for (int ms = 0; ms < 2; ++ms)
#pragma unroll
    for (int t = 0; t < 4; ++t) {
      const int n = n0 + 16 * t + c;
      const float bv = bias[n];
      const int hh = n >> 7, d = n & 127;
#pragma unroll
      for (int j = 0; j < 8; ++j) {
        const int m = m0 + ms * 16 + j + 8 * half;  // C layout: row = j + 8*half
        const int bidx = m >> SEQSHIFT;
        const int sl = m & ((1 << SEQSHIFT) - 1);
        const float val = acc[ms][t][j] + bv;
        size_t idx;
        if (MODE == 2)
          idx = ((size_t)(bidx * NH + hh) * DH + d) << SEQSHIFT, idx += sl;
        else
          idx = (((size_t)(bidx * NH + hh) << SEQSHIFT) + sl) * (size_t)DH + d;
        out[idx] = f2bf(val);
      }
    }
}

// ---------------------------------------------------------------------------
// Kernel 4: flash attention + residual.
// grid (QL/64, H, B), 128 threads = 4 waves, each wave: 16 q-rows x full 128 d.
// K/V tiles (32 keys) staged to LDS via async global->LDS, double-buffered,
// shared by all 4 waves.  q:[B*H,QL,DH] k:[B*H,KL,DH] vT:[B*H,DH,KL] bf16.
// Fragment loads are hoisted ahead of their consumers so LDS latency overlaps
// WMMA issue (scores) and the softmax VALU chain (ctx).
// ---------------------------------------------------------------------------
#define KT_STRIDE 136  // 32x128 k-tile, +8 pad: 272B rows (16B-aligned, no bank conflicts)
#define VT_STRIDE 40   // 128x32 vT-tile, +8 pad: 80B rows

__global__ __launch_bounds__(128) void attn_kernel(
    const unsigned short* __restrict__ qb, const unsigned short* __restrict__ kb,
    const unsigned short* __restrict__ vtb, const float* __restrict__ hidden,
    float* __restrict__ out) {
  const int tid  = threadIdx.x;
  const int lane = tid & 31;
  const int wave = tid >> 5;  // 0..3
  const int half = lane >> 4;
  const int c    = lane & 15;
  const int b = blockIdx.z, hh = blockIdx.y;
  const int bh = b * NH + hh;
  const int q0 = blockIdx.x * 64 + wave * 16;

  __shared__ unsigned short kt[2][32 * KT_STRIDE];
  __shared__ unsigned short vt[2][128 * VT_STRIDE];
  __shared__ unsigned short pb[4][16 * 32];  // per-wave P staging (C->A relayout)

  const unsigned short* kbase = kb + (size_t)bh * KLEN * DH;
  const unsigned short* vbase = vtb + (size_t)bh * DH * KLEN;

  // Preload q fragments: 16 rows x 128 d -> 4 A-fragments of 16x32
  FragA aq[4];
  {
    const unsigned short* qr = qb + ((size_t)bh * QL + q0 + c) * DH + half * 8;
#pragma unroll
    for (int f = 0; f < 4; ++f) {
      aq[f].h[0] = *(const v8bf*)(qr + f * 32);
      aq[f].h[1] = *(const v8bf*)(qr + f * 32 + 16);
    }
  }

  float mrow[8], lrow[8];
  v8f ctx[8];
#pragma unroll
  for (int r = 0; r < 8; ++r) { mrow[r] = -3.0e38f; lrow[r] = 0.f; }
#pragma unroll
  for (int t = 0; t < 8; ++t)
#pragma unroll
    for (int j = 0; j < 8; ++j) ctx[t][j] = 0.f;

  const float scale = 0.08838834764831845f;  // 1/sqrt(128)

  // Cooperative tile fill: 8 x b128 per thread (4 for K-tile, 4 for V-tile)
  auto issue_tiles = [&](int buf, int kpos) {
#pragma unroll
    for (int j = 0; j < 4; ++j) {
      const int cid = j * 128 + tid;                 // 0..511
      const int krow = cid >> 4, kcol = (cid & 15) * 8;
      async_cp16(kt[buf] + krow * KT_STRIDE + kcol,
                 kbase + (size_t)(kpos + krow) * DH + kcol);
      const int vrow = cid >> 2, vcol = (cid & 3) * 8;
      async_cp16(vt[buf] + vrow * VT_STRIDE + vcol,
                 vbase + (size_t)vrow * KLEN + kpos + vcol);
    }
  };

  issue_tiles(0, 0);
  int buf = 0;

  for (int kpos = 0; kpos < KLEN; kpos += 32) {
    const bool more = (kpos + 32) < KLEN;
    if (more) {
      issue_tiles(buf ^ 1, kpos + 32);
      wait_asynccnt<8>();   // first 8 (current buffer) complete, next 8 in flight
    } else {
      wait_asynccnt<0>();
    }
    __syncthreads();

    // ---- preload all 8 K B-fragments (clause of 16 ds_load_b128)
    FragA bk[8];
#pragma unroll
    for (int f = 0; f < 4; ++f) {
      const unsigned short* kp0 = kt[buf] + c * KT_STRIDE + f * 32 + half * 8;
      bk[f].h[0]     = *(const v8bf*)kp0;
      bk[f].h[1]     = *(const v8bf*)(kp0 + 16);
      const unsigned short* kp1 = kp0 + 16 * KT_STRIDE;
      bk[4 + f].h[0] = *(const v8bf*)kp1;
      bk[4 + f].h[1] = *(const v8bf*)(kp1 + 16);
    }

    // ---- S = q . k^T for 32 keys -> two 16x16 f32 tiles
    v8f s0, s1;
#pragma unroll
    for (int j = 0; j < 8; ++j) { s0[j] = 0.f; s1[j] = 0.f; }
#pragma unroll
    for (int f = 0; f < 4; ++f) {
      s0 = wmma_bf16(aq[f], bk[f], s0);
      s1 = wmma_bf16(aq[f], bk[4 + f], s1);
    }

    // ---- issue V B-fragment loads now; softmax VALU hides their latency
    FragA bv[8];
#pragma unroll
    for (int t = 0; t < 8; ++t) {
      const unsigned short* vp = vt[buf] + (t * 16 + c) * VT_STRIDE + half * 8;
      bv[t].h[0] = *(const v8bf*)vp;
      bv[t].h[1] = *(const v8bf*)(vp + 16);
    }

    // ---- online softmax (per-row stats live in 16-lane halves)
#pragma unroll
    for (int r = 0; r < 8; ++r) {
      float p0 = s0[r] * scale;
      float p1 = s1[r] * scale;
      float mx = fmaxf(p0, p1);
#pragma unroll
      for (int off = 8; off; off >>= 1) mx = fmaxf(mx, __shfl_xor(mx, off, 32));
      const float mnew = fmaxf(mrow[r], mx);
      const float corr = __expf(mrow[r] - mnew);
      mrow[r] = mnew;
      p0 = __expf(p0 - mnew);
      p1 = __expf(p1 - mnew);
      float ps = p0 + p1;
#pragma unroll
      for (int off = 8; off; off >>= 1) ps += __shfl_xor(ps, off, 32);
      lrow[r] = lrow[r] * corr + ps;
#pragma unroll
      for (int t = 0; t < 8; ++t) ctx[t][r] *= corr;
      const int rr = r + 8 * half;            // C-layout row
      pb[wave][rr * 32 + c]      = f2bf(p0);
      pb[wave][rr * 32 + 16 + c] = f2bf(p1);
    }

    // ---- reload P as an A-fragment (16 rows x 32 keys) from LDS
    FragA ap;
    ap.h[0] = *(const v8bf*)&pb[wave][c * 32 + half * 8];
    ap.h[1] = *(const v8bf*)&pb[wave][c * 32 + 16 + half * 8];

    // ---- ctx += P @ V
#pragma unroll
    for (int t = 0; t < 8; ++t) ctx[t] = wmma_bf16(ap, bv[t], ctx[t]);

    __syncthreads();   // all waves done with buf before it is refilled
    buf ^= 1;
  }

  // ---- normalize, add residual, write f32 output
#pragma unroll
  for (int t = 0; t < 8; ++t) {
    const int d = hh * DH + t * 16 + c;
#pragma unroll
    for (int r = 0; r < 8; ++r) {
      const int row = q0 + r + 8 * half;
      const size_t idx = ((size_t)b * QL + row) * DMODEL + d;
      out[idx] = hidden[idx] + ctx[t][r] / lrow[r];
    }
  }
}

// ---------------------------------------------------------------------------
extern "C" void kernel_launch(void* const* d_in, const int* in_sizes, int n_in,
                              void* d_out, int out_size, void* d_ws, size_t ws_size,
                              hipStream_t stream) {
  (void)in_sizes; (void)n_in; (void)out_size; (void)ws_size;
  const float* hidden = (const float*)d_in[0];
  const float* inputs = (const float*)d_in[1];
  const float* ln1w = (const float*)d_in[2];
  const float* ln1b = (const float*)d_in[3];
  const float* ln2w = (const float*)d_in[4];
  const float* ln2b = (const float*)d_in[5];
  const float* Wq = (const float*)d_in[6];
  const float* bq = (const float*)d_in[7];
  const float* Wk = (const float*)d_in[8];
  const float* bk = (const float*)d_in[9];
  const float* Wv = (const float*)d_in[10];
  const float* bv = (const float*)d_in[11];

  // workspace layout (bf16 elements), ~224 MB total
  unsigned short* ws   = (unsigned short*)d_ws;
  unsigned short* hln  = ws;
  unsigned short* xln  = hln  + (size_t)BB * QL * DMODEL;
  unsigned short* wqb  = xln  + (size_t)BB * KLEN * DMODEL;
  unsigned short* wkb  = wqb  + (size_t)DMODEL * DMODEL;
  unsigned short* wvb  = wkb  + (size_t)DMODEL * DMODEL;
  unsigned short* qbuf = wvb  + (size_t)DMODEL * DMODEL;
  unsigned short* kbuf = qbuf + (size_t)BB * QL * DMODEL;
  unsigned short* vtbf = kbuf + (size_t)BB * KLEN * DMODEL;

  ln_bf16_kernel<<<BB * QL, 256, 0, stream>>>(hidden, ln1w, ln1b, hln);
  ln_bf16_kernel<<<BB * KLEN, 256, 0, stream>>>(inputs, ln2w, ln2b, xln);

  const int wn = DMODEL * DMODEL;
  f32_to_bf16_kernel<<<wn / 256, 256, 0, stream>>>(Wq, wqb, wn);
  f32_to_bf16_kernel<<<wn / 256, 256, 0, stream>>>(Wk, wkb, wn);
  f32_to_bf16_kernel<<<wn / 256, 256, 0, stream>>>(Wv, wvb, wn);

  proj_gemm_kernel<9, 0><<<dim3((BB * QL) / 128, DMODEL / 128), 256, 0, stream>>>(
      hln, wqb, bq, qbuf);
  proj_gemm_kernel<12, 1><<<dim3((BB * KLEN) / 128, DMODEL / 128), 256, 0, stream>>>(
      xln, wkb, bk, kbuf);
  proj_gemm_kernel<12, 2><<<dim3((BB * KLEN) / 128, DMODEL / 128), 256, 0, stream>>>(
      xln, wvb, bv, vtbf);

  attn_kernel<<<dim3(QL / 64, NH, BB), 128, 0, stream>>>(
      qbuf, kbuf, vtbf, hidden, (float*)d_out);
}
